// FFTConv2D_39496519254674
// MI455X (gfx1250) — matrix operand order repaired and checked by reading, not compiled, not run on hardware
//
#include <hip/hip_runtime.h>

// ---------------------------------------------------------------------------
// FFT-conv reference == valid true convolution:
//   y[b,o,r,c] = sum_{i,u,v} x[b,i,r+u,c+v] * w[o,i,6-u,6-v] + bias[o]
// Implicit GEMM on the CDNA5 matrix pipe:
//   M = Cout = 128, N = 16*122*122, K = 3136, K-order k = (u*7+v)*64 + ci
//   => each 32-wide k-step has fixed (u,v), ci-linear: loader is strided loads.
// bf16 inputs, f32 accumulation via v_wmma_f32_16x16x32_bf16.
// A panel staged global->LDS with TENSOR_LOAD_TO_LDS (TDM, TENSORcnt) or
// global_load_async_to_lds_b128 (ASYNCcnt) fallback; double-buffered.
// ---------------------------------------------------------------------------

typedef __bf16  v16bf __attribute__((ext_vector_type(16)));
typedef float   v8f   __attribute__((ext_vector_type(8)));
typedef unsigned int v4u __attribute__((ext_vector_type(4)));
typedef int          v8i __attribute__((ext_vector_type(8)));
typedef int          v4i __attribute__((ext_vector_type(4)));

#if defined(__AMDGCN__) && __has_builtin(__builtin_amdgcn_tensor_load_to_lds)
#define HAVE_TDM 1
#else
#define HAVE_TDM 0
#endif

static constexpr int B_    = 16;
static constexpr int CIN   = 64;
static constexpr int H_    = 128;
static constexpr int W_    = 128;
static constexpr int HW_   = H_ * W_;          // 16384
static constexpr int COUT  = 128;
static constexpr int HO    = 122;
static constexpr int WO    = 122;
static constexpr int HWO   = HO * WO;          // 14884
static constexpr int KTOT  = CIN * 49;         // 3136
static constexpr int KSTEP = 32;
static constexpr int NSTEP = KTOT / KSTEP;     // 98
static constexpr int NBLK  = 128;              // 2 rows x 64 contiguous cols per block
static constexpr int ROWB  = 80;               // LDS bytes per B n-row (64B data + 16B pad)
static constexpr int AROWB = 80;               // LDS bytes per A o-row (64B data + 16B pad)
static constexpr size_t WPK_BYTES = (size_t)COUT * KTOT * 2;        // 802816
static constexpr size_t XBF_BYTES = (size_t)B_ * CIN * H_ * W_ * 2; // 33554432

__device__ __forceinline__ unsigned short f2bf(float f) {
    union { float f; unsigned int u; } v; v.f = f;
    unsigned int u = v.u;
    u += 0x7FFFu + ((u >> 16) & 1u);           // round-to-nearest-even
    return (unsigned short)(u >> 16);
}

union AFrag { uint4 q[2]; v16bf v; };
union BFrag { uint4 q[2]; v16bf v; };

// --- prepass 1: wpk[o][(u*7+v)*64+ci] = bf16( w[o,ci,6-u,6-v] ) -------------
__global__ void pack_weights_kernel(const float* __restrict__ w,
                                    unsigned short* __restrict__ wpk) {
    int idx = blockIdx.x * 256 + threadIdx.x;  // 128*3136 = 401408 exact
    if (idx >= COUT * KTOT) return;
    int o  = idx / KTOT;
    int k  = idx - o * KTOT;
    int uv = k >> 6;
    int ci = k & 63;
    int u  = uv / 7;
    int v  = uv - u * 7;
    wpk[idx] = f2bf(w[((o * CIN + ci) * 7 + (6 - u)) * 7 + (6 - v)]);
}

// --- prepass 2: x (f32) -> xbf (bf16), 4 elements / thread ------------------
__global__ void convert_x_kernel(const float* __restrict__ x,
                                 unsigned short* __restrict__ xb) {
    int i = (blockIdx.x * 256 + threadIdx.x) * 4;   // total 16*64*128*128 exact
    float4 f = *(const float4*)(x + i);
    uint2 p;
    p.x = (unsigned)f2bf(f.x) | ((unsigned)f2bf(f.y) << 16);
    p.y = (unsigned)f2bf(f.z) | ((unsigned)f2bf(f.w) << 16);
    *(uint2*)(xb + i) = p;
}

// --- main: 128(Cout) x 128(pos) macro-tile per block ------------------------
template <bool BF16X>
__global__ void __launch_bounds__(256)
fftconv_wmma_kernel(const void* __restrict__ xsrc,
                    const unsigned short* __restrict__ wpk,
                    const float* __restrict__ bias,
                    float* __restrict__ y) {
    __shared__ __align__(16) unsigned char As[2 * COUT * AROWB];  // 20480 B
    __shared__ __align__(16) unsigned char Bs[2 * NBLK * ROWB];   // 20480 B

    const int t    = threadIdx.x;
    const int lane = t & 31;
    const int wid  = t >> 5;                  // wave 0..7 -> Cout tile
    const int mi   = lane & 15;
    const int h    = lane >> 4;
    const int m_base = wid * 16;

    const int bid  = blockIdx.x;              // 16 * 61 * 2 = 1952 blocks
    const int cblk = bid & 1;
    const int rp   = (bid >> 1) % 61;
    const int b    = bid / 122;
    const int r0   = rp * 2;
    const int c_base = cblk * 58;             // {0,58}: cols 58..63 overlap, identical recompute

    // ---- B-loader role: thread owns column nl, k-rows [16*kg, 16*kg+16) ----
    const int nl   = t & 127;
    const int kg   = t >> 7;                  // 0/1
    const int rowh = nl >> 6;                 // which of the 2 output rows
    const int nlc  = nl & 63;
    const size_t xoff = ((size_t)(b * CIN) * H_ + (r0 + rowh)) * W_ + c_base + nlc;
    const unsigned short* xb16 = (const unsigned short*)xsrc + xoff;
    const float*          xb32 = (const float*)xsrc + xoff;

    // uniform per-step loader state: k-step s -> uv = s>>1, ci0 = 32*(s&1)
    int v_l = 0, half_l = 0, uvoff = 0;       // uvoff = u*W_ + v

    auto loadB = [&](int buf) {
        unsigned char* dst = Bs + buf * (NBLK * ROWB) + nl * ROWB + kg * 32;
        const int cibase = half_l * 32 + kg * 16;
        if constexpr (BF16X) {
            const unsigned short* p = xb16 + (size_t)cibase * HW_ + uvoff;
            #pragma unroll
            for (int q = 0; q < 8; ++q) {
                unsigned lo = p[(2 * q)     * HW_];
                unsigned hi = p[(2 * q + 1) * HW_];
                *(unsigned int*)(dst + q * 4) = lo | (hi << 16);
            }
        } else {
            const float* p = xb32 + (size_t)cibase * HW_ + uvoff;
            #pragma unroll
            for (int q = 0; q < 8; ++q) {
                unsigned lo = f2bf(p[(2 * q)     * HW_]);
                unsigned hi = f2bf(p[(2 * q + 1) * HW_]);
                *(unsigned int*)(dst + q * 4) = lo | (hi << 16);
            }
        }
        if (++half_l == 2) {                  // advance (u,v) once per 2 steps
            half_l = 0;
            ++uvoff;
            if (++v_l == 7) { v_l = 0; uvoff += W_ - 7; }
        }
    };

    // ---- A-loader: 8KB weight k-panel -> LDS, padded 64B rows -> 80B -------
#if !HAVE_TDM
    const int ao = t >> 1;                    // o row 0..127
    const int ac = (t & 1) * 2;               // chunk pair {ac, ac+1} of 4
#endif
    auto loadA = [&](int buf, int k0) {
#if HAVE_TDM
        if (wid == 0) {                       // one wave drives the TDM
            unsigned lds = (unsigned)(uintptr_t)(As + buf * (COUT * AROWB));
            unsigned long long ga = (unsigned long long)(uintptr_t)(wpk + k0);
            // D# group0: count=1 | lds_addr | global_addr | type=2
            v4u g0 = { 1u, lds, (unsigned)ga,
                       (unsigned)((ga >> 32) & 0x01FFFFFFu) | 0x80000000u };
            // D# group1: data_size=2B, pad_enable, pad_interval=64B, pad_amount=16B,
            //            tensor_dim0=32, tensor_dim1=128, tile_dim0=32, tile_dim1=128,
            //            tensor_dim0_stride=3136
            v8i g1 = { (int)0x06D10000u,
                       (int)(32u << 16),      // tensor_dim0 = 32
                       (int)(128u << 16),     // tensor_dim1 = 128
                       (int)(32u << 16),      // tile_dim0 = 32
                       128,                   // tile_dim1 = 128
                       KTOT,                  // tensor_dim0_stride
                       0, 0 };
            v4i gz4 = { 0, 0, 0, 0 };
            v8i gz8 = { 0, 0, 0, 0, 0, 0, 0, 0 };
            __builtin_amdgcn_tensor_load_to_lds(g0, g1, gz4, gz4, gz8, 0);
        }
#else
        unsigned lds0 = (unsigned)(uintptr_t)(As + buf * (COUT * AROWB) + ao * AROWB + ac * 16);
        unsigned long long ga0 =
            (unsigned long long)(uintptr_t)(wpk + (size_t)ao * KTOT + k0 + ac * 8);
        asm volatile("global_load_async_to_lds_b128 %0, %1, off"
                     :: "v"(lds0), "v"(ga0) : "memory");
        unsigned lds1 = lds0 + 16;
        unsigned long long ga1 = ga0 + 16;
        asm volatile("global_load_async_to_lds_b128 %0, %1, off"
                     :: "v"(lds1), "v"(ga1) : "memory");
#endif
    };

    auto waitA = [&]() {
#if HAVE_TDM
        if (wid == 0) __builtin_amdgcn_s_wait_tensorcnt(0);
#else
        asm volatile("s_wait_asynccnt 0x0" ::: "memory");
#endif
    };

    v8f acc[8];
    #pragma unroll
    for (int i = 0; i < 8; ++i) acc[i] = v8f{};

    // ---- prologue: stage tile 0 ----
    loadA(0, 0);
    loadB(0);
    waitA();
    __syncthreads();

    // ---- main loop: double-buffered, 98 k-steps ----
    for (int s = 0; s < NSTEP; ++s) {
        const int cur = s & 1;
        if (s + 1 < NSTEP) {                  // overlap next tile with compute
            loadA(cur ^ 1, (s + 1) * KSTEP);
            loadB(cur ^ 1);
        }

        // A fragment: lane(m,h) -> K run [16h,16h+16)
        AFrag a;
        const unsigned char* ar = As + cur * (COUT * AROWB) + (m_base + mi) * AROWB;
        a.q[0] = *(const uint4*)(ar + 32 * h);
        a.q[1] = *(const uint4*)(ar + 32 * h + 16);

        #pragma unroll
        for (int nt = 0; nt < 8; ++nt) {
            // B fragment: lane(n=mi,h) -> K run [16h,16h+16) of column nt*16+mi
            const unsigned char* br =
                Bs + cur * (NBLK * ROWB) + (nt * 16 + mi) * ROWB + 32 * h;
            BFrag bf;
            bf.q[0] = *(const uint4*)(br);
            bf.q[1] = *(const uint4*)(br + 16);
            acc[nt] = __builtin_amdgcn_wmma_f32_16x16x32_bf16(
                false, a.v, false, bf.v, (short)0, acc[nt], false, false);
        }

        waitA();
        __syncthreads();
    }

    // ---- epilogue: +bias, non-temporal scatter (C layout: comp i -> M=i+8h) ----
    float bi[8];
    {
        float4 b0 = *(const float4*)(bias + m_base + 8 * h);
        float4 b1 = *(const float4*)(bias + m_base + 8 * h + 4);
        bi[0]=b0.x; bi[1]=b0.y; bi[2]=b0.z; bi[3]=b0.w;
        bi[4]=b1.x; bi[5]=b1.y; bi[6]=b1.z; bi[7]=b1.w;
    }
    #pragma unroll
    for (int nt = 0; nt < 8; ++nt) {
        int n  = nt * 16 + mi;
        int rh = n >> 6;
        int c  = c_base + (n & 63);
        float* op = y + (size_t)b * COUT * HWO + (size_t)(r0 + rh) * WO + c;
        #pragma unroll
        for (int i = 0; i < 8; ++i) {
            int o = m_base + i + 8 * h;
            __builtin_nontemporal_store(acc[nt][i] + bi[i], op + (size_t)o * HWO);
        }
    }
}

// ---------------------------------------------------------------------------
extern "C" void kernel_launch(void* const* d_in, const int* in_sizes, int n_in,
                              void* d_out, int out_size, void* d_ws, size_t ws_size,
                              hipStream_t stream) {
    const float* x    = (const float*)d_in[0];
    const float* w    = (const float*)d_in[1];
    const float* bias = (const float*)d_in[2];
    float*       y    = (float*)d_out;
    unsigned short* wpk = (unsigned short*)d_ws;

    pack_weights_kernel<<<(COUT * KTOT) / 256, 256, 0, stream>>>(w, wpk);

    const int nblocks = 16 * 61 * 2;          // 1952
    if (ws_size >= WPK_BYTES + XBF_BYTES) {
        unsigned short* xbf = (unsigned short*)((char*)d_ws + WPK_BYTES);
        convert_x_kernel<<<(B_ * CIN * H_ * W_ / 4) / 256, 256, 0, stream>>>(x, xbf);
        fftconv_wmma_kernel<true><<<nblocks, 256, 0, stream>>>(xbf, wpk, bias, y);
    } else {
        fftconv_wmma_kernel<false><<<nblocks, 256, 0, stream>>>(x, wpk, bias, y);
    }
}
